// TransformerBlock_36481452212932
// MI455X (gfx1250) — matrix operand order repaired and checked by reading, not compiled, not run on hardware
//
#include <hip/hip_runtime.h>
#include <hip/hip_bf16.h>
#include <math.h>

#define Bsz 2
#define Ssz 1024
#define Dsz 1536
#define HQn 24
#define HKn 3
#define HDd 64
#define En  16
#define KTOP 4
#define Fn  768
#define Tn  (Bsz*Ssz)
#define EPSv 1e-5f

typedef __attribute__((ext_vector_type(16))) __bf16 v16bf;
typedef __attribute__((ext_vector_type(8)))  __bf16 v8bf;
typedef __attribute__((ext_vector_type(8)))  float  v8f;
typedef __attribute__((ext_vector_type(4)))  unsigned int u32x4;
typedef __attribute__((ext_vector_type(8)))  int i32x8;
typedef __attribute__((ext_vector_type(4)))  int i32x4;

#if defined(__has_builtin)
#if __has_builtin(__builtin_amdgcn_tensor_load_to_lds) && __has_builtin(__builtin_amdgcn_s_wait_tensorcnt)
#define HAVE_TDM 1
#endif
#endif

#define DI __device__ __forceinline__

DI v8f wmma_bf16(v16bf a, v16bf b, v8f c) {
  return __builtin_amdgcn_wmma_f32_16x16x32_bf16(false, a, false, b, (short)0, c, false, false);
}

// A-matrix 16x32 bf16 loader. base = &A[mBase*ld + kBase], row-major, ld elems.
// Lane L: row L&15. K = (L>>4)*16 + 0..15.
DI v16bf load_A(const __bf16* base, int ld, int lane) {
  const int r = lane & 15, h = lane >> 4;
  const __bf16* rp = base + (size_t)r * ld;
  v8bf a0 = *(const v8bf*)(rp + h * 16);
  v8bf a1 = *(const v8bf*)(rp + h * 16 + 8);
  v16bf a;
#pragma unroll
  for (int i = 0; i < 8; i++) { a[i] = a0[i]; a[8 + i] = a1[i]; }
  return a;
}

// B 32x16 from bf16 where N maps to memory rows (K contiguous within a row).
// base = &W[nBase*ld + kBase]. Lane L: col L&15, K = (L>>4)*16 + 0..15.
DI v16bf load_B_bf16_rows(const __bf16* base, int ld, int lane) {
  const __bf16* p = base + (size_t)(lane & 15) * ld + ((lane >> 4) * 16);
  v8bf b0 = *(const v8bf*)(p);
  v8bf b1 = *(const v8bf*)(p + 8);
  v16bf b;
#pragma unroll
  for (int i = 0; i < 8; i++) { b[i] = b0[i]; b[8 + i] = b1[i]; }
  return b;
}

// B 32x16 from bf16 where K maps to memory rows (strided column reads).
DI v16bf load_B_bf16_cols(const __bf16* base, int ld, int lane) {
  const __bf16* p = base + (size_t)((lane >> 4) * 16) * ld + (lane & 15);
  v16bf b;
#pragma unroll
  for (int i = 0; i < 16; i++) b[i] = p[(size_t)i * ld];
  return b;
}

// ---------------- fp32 -> bf16 weight conversion ----------------
__global__ void cvt_bf16_kernel(const float* __restrict__ src, __bf16* __restrict__ dst,
                                long n) {
  long i = (long)blockIdx.x * blockDim.x + threadIdx.x;
  const long stride = (long)gridDim.x * blockDim.x;
  for (; i < n; i += stride) dst[i] = (__bf16)src[i];
}

// ---------------- RMSNorm -> bf16 ----------------
__global__ void rmsnorm_kernel(const float* __restrict__ x, const float* __restrict__ w,
                               __bf16* __restrict__ out) {
  const int row = blockIdx.x;
  const int tid = threadIdx.x;
  __shared__ float red[256];
  const float* xr = x + (size_t)row * Dsz;
  float ss = 0.f;
  for (int i = tid; i < Dsz; i += 256) { float v = xr[i]; ss += v * v; }
  red[tid] = ss; __syncthreads();
  for (int s = 128; s > 0; s >>= 1) { if (tid < s) red[tid] += red[tid + s]; __syncthreads(); }
  const float rs = rsqrtf(red[0] / (float)Dsz + EPSv);
  __bf16* orow = out + (size_t)row * Dsz;
  for (int i = tid; i < Dsz; i += 256) orow[i] = (__bf16)(xr[i] * rs * w[i]);
}

// -------- Register-blocked WMMA GEMM: C[MxN] = A_bf16[MxK] * Wbf16[NxK]^T ----
// One wave computes a 32(M) x 64(N) tile: 2 A-frags x 4 B-frags -> 8 WMMAs/k.
// OUT==0: bf16 out; OUT==1: f32 out; OUT==2: f32 out + residual.
template<int OUT>
__global__ void gemm_kernel(const __bf16* __restrict__ A, const __bf16* __restrict__ W,
                            int N, int Kd, void* __restrict__ Cout,
                            const float* __restrict__ Res) {
  const int lane = threadIdx.x;
  const int nBase = blockIdx.x * 64, mBase = blockIdx.y * 32;
  v8f acc[2][4] = {};
  const __bf16* Ab0 = A + (size_t)mBase * Kd;
  const __bf16* Ab1 = Ab0 + (size_t)16 * Kd;
  const __bf16* Wb  = W + (size_t)nBase * Kd;
  for (int k = 0; k < Kd; k += 32) {
    __builtin_prefetch(Wb + (size_t)(lane & 15) * Kd + k + 256, 0, 1);
    v16bf a0 = load_A(Ab0 + k, Kd, lane);
    v16bf a1 = load_A(Ab1 + k, Kd, lane);
#pragma unroll
    for (int nt = 0; nt < 4; nt++) {
      v16bf b = load_B_bf16_rows(Wb + (size_t)(nt * 16) * Kd + k, Kd, lane);
      acc[0][nt] = wmma_bf16(a0, b, acc[0][nt]);
      acc[1][nt] = wmma_bf16(a1, b, acc[1][nt]);
    }
  }
#pragma unroll
  for (int mi = 0; mi < 2; mi++) {
    const int r0 = mBase + mi * 16 + 8 * (lane >> 4);
#pragma unroll
    for (int nt = 0; nt < 4; nt++) {
      const int col = nBase + nt * 16 + (lane & 15);
#pragma unroll
      for (int i = 0; i < 8; i++) {
        size_t o = (size_t)(r0 + i) * N + col;
        if (OUT == 0)      ((__bf16*)Cout)[o] = (__bf16)acc[mi][nt][i];
        else if (OUT == 1) ((float*)Cout)[o]  = acc[mi][nt][i];
        else               ((float*)Cout)[o]  = acc[mi][nt][i] + Res[o];
      }
    }
  }
}

// Small-N GEMM (one 16x16 tile per wave) for the 16-wide gate logits.
__global__ void gemm16_kernel(const __bf16* __restrict__ A, const __bf16* __restrict__ W,
                              int N, int Kd, float* __restrict__ Cout) {
  const int lane = threadIdx.x;
  const int nBase = blockIdx.x * 16, mBase = blockIdx.y * 16;
  v8f acc = {};
  const __bf16* Ab = A + (size_t)mBase * Kd;
  const __bf16* Wb = W + (size_t)nBase * Kd;
  for (int k = 0; k < Kd; k += 32) {
    v16bf a = load_A(Ab + k, Kd, lane);
    v16bf b = load_B_bf16_rows(Wb + k, Kd, lane);
    acc = wmma_bf16(a, b, acc);
  }
  const int col = nBase + (lane & 15);
  const int r0  = mBase + 8 * (lane >> 4);
#pragma unroll
  for (int i = 0; i < 8; i++) Cout[(size_t)(r0 + i) * N + col] = acc[i];
}

// ---------------- RoPE (in place on bf16, pre-repeat heads) -----------------
__global__ void rope_kernel(__bf16* __restrict__ t, int Hn_,
                            const float* __restrict__ cosT, const float* __restrict__ sinT) {
  const int i = blockIdx.x * blockDim.x + threadIdx.x;
  const int total = Tn * Hn_ * (HDd / 2);
  if (i >= total) return;
  const int p  = i & 31;
  const int hh = (i >> 5) % Hn_;
  const int tk = i / (32 * Hn_);
  const int s  = tk % Ssz;
  const float c = cosT[s * 32 + p], sn = sinT[s * 32 + p];
  __bf16* q = t + (size_t)tk * (Hn_ * HDd) + hh * HDd + 2 * p;
  const float e = (float)q[0], o = (float)q[1];
  q[0] = (__bf16)(e * c - o * sn);
  q[1] = (__bf16)(e * sn + o * c);
}

// ---------------- Flash attention (1 wave / 16-query tile) ------------------
__global__ void attn_kernel(const __bf16* __restrict__ Q, const __bf16* __restrict__ Kt,
                            const __bf16* __restrict__ Vt, __bf16* __restrict__ Ctx) {
  const int lane = threadIdx.x;
  const int qt = blockIdx.x, h = blockIdx.y, bb = blockIdx.z;
  const int kvh = h / (HQn / HKn);
  const int ldq = HQn * HDd, ldk = HKn * HDd;
  const size_t tq0 = (size_t)(bb * Ssz + qt * 16);
  const size_t tk0 = (size_t)(bb * Ssz);
  __shared__ __align__(16) __bf16 Plds[16 * 32];

  const v16bf aq0 = load_A(Q + tq0 * ldq + h * HDd + 0,  ldq, lane);
  const v16bf aq1 = load_A(Q + tq0 * ldq + h * HDd + 32, ldq, lane);

  v8f acc0 = {}, acc1 = {}, acc2 = {}, acc3 = {};
  float mrow[8], lrow[8];
#pragma unroll
  for (int i = 0; i < 8; i++) { mrow[i] = -INFINITY; lrow[i] = 0.f; }

  const int qEnd = qt * 16 + 15;
  for (int jb = 0; jb <= qEnd; jb += 32) {
    v8f s0 = {}, s1 = {};
    const __bf16* kb0 = Kt + (tk0 + jb)      * ldk + kvh * HDd;
    const __bf16* kb1 = Kt + (tk0 + jb + 16) * ldk + kvh * HDd;
    {
      v16bf b0 = load_B_bf16_rows(kb0 + 0, ldk, lane);
      v16bf b1 = load_B_bf16_rows(kb1 + 0, ldk, lane);
      s0 = wmma_bf16(aq0, b0, s0);
      s1 = wmma_bf16(aq0, b1, s1);
      b0 = load_B_bf16_rows(kb0 + 32, ldk, lane);
      b1 = load_B_bf16_rows(kb1 + 32, ldk, lane);
      s0 = wmma_bf16(aq1, b0, s0);
      s1 = wmma_bf16(aq1, b1, s1);
    }
    const int key0 = jb + (lane & 15);
    const int key1 = key0 + 16;
    float rowmax[8];
#pragma unroll
    for (int i = 0; i < 8; i++) {
      const int qrow = qt * 16 + 8 * (lane >> 4) + i;
      const float v0 = (key0 <= qrow) ? s0[i] * 0.125f : -INFINITY;
      const float v1 = (key1 <= qrow) ? s1[i] * 0.125f : -INFINITY;
      s0[i] = v0; s1[i] = v1;
      rowmax[i] = fmaxf(v0, v1);
    }
#pragma unroll
    for (int i = 0; i < 8; i++) {
      float v = rowmax[i];
      for (int off = 1; off < 16; off <<= 1) v = fmaxf(v, __shfl_xor(v, off, 32));
      rowmax[i] = v;
    }
    float p0[8], p1[8];
#pragma unroll
    for (int i = 0; i < 8; i++) {
      const float mn = fmaxf(mrow[i], rowmax[i]);
      const float sc = __expf(mrow[i] - mn);
      mrow[i] = mn;
      p0[i] = __expf(s0[i] - mn);
      p1[i] = __expf(s1[i] - mn);
      float rs = p0[i] + p1[i];
      for (int off = 1; off < 16; off <<= 1) rs += __shfl_xor(rs, off, 32);
      lrow[i] = lrow[i] * sc + rs;
      acc0[i] *= sc; acc1[i] *= sc; acc2[i] *= sc; acc3[i] *= sc;
    }
    {
      const int c = lane & 15, hb = lane >> 4;
#pragma unroll
      for (int i = 0; i < 8; i++) {
        const int r = 8 * hb + i;
        Plds[r * 32 + c]      = (__bf16)p0[i];
        Plds[r * 32 + 16 + c] = (__bf16)p1[i];
      }
    }
    __syncthreads();
    const v16bf ap = load_A(Plds, 32, lane);
    const __bf16* vbp = Vt + (tk0 + jb) * ldk + kvh * HDd;
    acc0 = wmma_bf16(ap, load_B_bf16_cols(vbp + 0,  ldk, lane), acc0);
    acc1 = wmma_bf16(ap, load_B_bf16_cols(vbp + 16, ldk, lane), acc1);
    acc2 = wmma_bf16(ap, load_B_bf16_cols(vbp + 32, ldk, lane), acc2);
    acc3 = wmma_bf16(ap, load_B_bf16_cols(vbp + 48, ldk, lane), acc3);
    __syncthreads();
  }
  const int c = lane & 15;
#pragma unroll
  for (int i = 0; i < 8; i++) {
    const int r = 8 * (lane >> 4) + i;
    const float inv = 1.f / lrow[i];
    const size_t o = (tq0 + r) * (size_t)(HQn * HDd) + h * HDd;
    Ctx[o + c]      = (__bf16)(acc0[i] * inv);
    Ctx[o + 16 + c] = (__bf16)(acc1[i] * inv);
    Ctx[o + 32 + c] = (__bf16)(acc2[i] * inv);
    Ctx[o + 48 + c] = (__bf16)(acc3[i] * inv);
  }
}

// ---------------- Gating: softmax + deterministic top-4 ---------------------
__global__ void gate_topk_kernel(const float* __restrict__ logits, int* __restrict__ tk_idx,
                                 float* __restrict__ tk_wt) {
  const int t = blockIdx.x * blockDim.x + threadIdx.x;
  if (t >= Tn) return;
  float lg[En], pr[En], comp[En];
  float mx = -INFINITY;
  for (int e = 0; e < En; e++) { lg[e] = logits[t * En + e]; mx = fmaxf(mx, lg[e]); }
  float sum = 0.f;
  for (int e = 0; e < En; e++) { pr[e] = __expf(lg[e] - mx); sum += pr[e]; }
  const float inv = 1.f / sum;
  for (int e = 0; e < En; e++) { pr[e] *= inv; comp[e] = pr[e] - (float)e * 1e-6f; }
  float wsum = 0.f;
  int idxs[KTOP]; float vals[KTOP];
  for (int k = 0; k < KTOP; k++) {
    int bi = 0; float bv = -INFINITY;
    for (int e = 0; e < En; e++) { if (comp[e] > bv) { bv = comp[e]; bi = e; } }
    idxs[k] = bi; vals[k] = pr[bi]; comp[bi] = -INFINITY; wsum += pr[bi];
  }
  const float wn = 1.f / wsum;
  for (int k = 0; k < KTOP; k++) { tk_idx[t * KTOP + k] = idxs[k]; tk_wt[t * KTOP + k] = vals[k] * wn; }
}

// ---------------- Deterministic per-expert compaction -----------------------
__global__ void compact_kernel(const int* __restrict__ tk_idx, const float* __restrict__ tk_wt,
                               int* __restrict__ counts, int* __restrict__ tok,
                               float* __restrict__ wtl) {
  const int e = threadIdx.x;
  if (e >= En) return;
  int c = 0;
  for (int t = 0; t < Tn; t++)
    for (int k = 0; k < KTOP; k++)
      if (tk_idx[t * KTOP + k] == e) {
        tok[e * Tn + c] = t; wtl[e * Tn + c] = tk_wt[t * KTOP + k]; c++;
      }
  counts[e] = c;
}

// ---------------- MoE: TDM-gathered tokens, SwiGLU, down-proj, scatter ------
__global__ void moe_kernel(const __bf16* __restrict__ hn, const __bf16* __restrict__ Wg,
                           const __bf16* __restrict__ Wu, const __bf16* __restrict__ Wd,
                           const int* __restrict__ counts, const int* __restrict__ tok,
                           const float* __restrict__ wtl, float* __restrict__ Y) {
  const int e = blockIdx.y;
  const int tile = blockIdx.x;
  const int cnt = counts[e];
  if (tile * 16 >= cnt) return;
  __shared__ int   sTok[16];
  __shared__ float sWt[16];
  __shared__ __align__(16) __bf16 hnTile[16 * Dsz];  // 48 KB (TDM gather dest)
  __shared__ __align__(16) __bf16 act[16 * Fn];      // 24 KB
  const int tid = threadIdx.x;
  if (tid < 16) {
    const int li = tile * 16 + tid;
    const int cl = li < cnt ? li : cnt - 1;
    sTok[tid] = tok[e * Tn + cl];
    sWt[tid]  = (li < cnt) ? wtl[e * Tn + li] : 0.f;
  }
  __syncthreads();

#ifdef HAVE_TDM
  if (tid < 32) {
    // Tensor DMA gather: 16 rows of hn (16-bit row indices) -> hnTile in LDS.
    const unsigned lds_off = (unsigned)(size_t)(&hnTile[0]);
    const unsigned long long ga = (unsigned long long)(size_t)hn;
    u32x4 g0;
    g0[0] = 0x80000001u;                       // gather_mode=1, 16b idx, count=1
    g0[1] = lds_off;                           // lds_addr
    g0[2] = (unsigned)(ga & 0xFFFFFFFFu);      // global_addr[31:0]
    g0[3] = (unsigned)((ga >> 32) & 0x1FFFFFFu) | (2u << 30);  // addr hi | type=2
    i32x8 g1;
    g1[0] = (int)(1u << 16);                   // wg_mask=0, data_size=1 (2B)
    g1[1] = (int)((unsigned)(Dsz & 0xFFFF) << 16);   // tensor_dim0[15:0] @ bit48
    g1[2] = (int)(((Dsz >> 16) & 0xFFFF) | ((unsigned)(Tn & 0xFFFF) << 16));
    g1[3] = (int)(((Tn >> 16) & 0xFFFF) | ((unsigned)(Dsz & 0xFFFF) << 16));  // tile_dim0
    g1[4] = 16;                                // tile_dim1 = #valid indices
    g1[5] = Dsz;                               // tensor_dim0_stride[31:0]
    g1[6] = 0;
    g1[7] = 0;
    i32x4 g2, g3;
#pragma unroll
    for (int w = 0; w < 4; w++) {
      g2[w] = (sTok[2 * w] & 0xFFFF)     | (sTok[2 * w + 1] << 16);
      g3[w] = (sTok[8 + 2 * w] & 0xFFFF) | (sTok[9 + 2 * w] << 16);
    }
#if defined(__clang_major__) && (__clang_major__ >= 23)
    i32x8 gz = {};
    __builtin_amdgcn_tensor_load_to_lds(g0, g1, g2, g3, gz, 0);
#else
    __builtin_amdgcn_tensor_load_to_lds(g0, g1, g2, g3, 0);
#endif
    __builtin_amdgcn_s_wait_tensorcnt(0);
  }
  __syncthreads();
#else
  for (int i = tid; i < 16 * (Dsz / 8); i += 256) {
    const int r = i / (Dsz / 8), cb = i % (Dsz / 8);
    ((v8bf*)hnTile)[r * (Dsz / 8) + cb] =
        ((const v8bf*)(hn + (size_t)sTok[r] * Dsz))[cb];
  }
  __syncthreads();
#endif

  const int wave = tid >> 5, lane = tid & 31;
  const __bf16* WgE = Wg + (size_t)e * Fn * Dsz;
  const __bf16* WuE = Wu + (size_t)e * Fn * Dsz;
  const __bf16* WdE = Wd + (size_t)e * Dsz * Fn;

  // Phase 1: gate/up GEMMs over F (48 n-tiles, 6 per wave, paired for A reuse)
  for (int np = 0; np < 3; np++) {
    const int nb0 = (wave * 6 + np * 2) * 16;
    const int nb1 = nb0 + 16;
    v8f ag0 = {}, au0 = {}, ag1 = {}, au1 = {};
    for (int k = 0; k < Dsz; k += 32) {
      v16bf a   = load_A(hnTile + k, Dsz, lane);
      v16bf bg0 = load_B_bf16_rows(WgE + (size_t)nb0 * Dsz + k, Dsz, lane);
      v16bf bu0 = load_B_bf16_rows(WuE + (size_t)nb0 * Dsz + k, Dsz, lane);
      v16bf bg1 = load_B_bf16_rows(WgE + (size_t)nb1 * Dsz + k, Dsz, lane);
      v16bf bu1 = load_B_bf16_rows(WuE + (size_t)nb1 * Dsz + k, Dsz, lane);
      ag0 = wmma_bf16(a, bg0, ag0);
      au0 = wmma_bf16(a, bu0, au0);
      ag1 = wmma_bf16(a, bg1, ag1);
      au1 = wmma_bf16(a, bu1, au1);
    }
    const int c = lane & 15;
#pragma unroll
    for (int i = 0; i < 8; i++) {
      const int r = 8 * (lane >> 4) + i;
      const float g0v = ag0[i];
      const float g1v = ag1[i];
      act[r * Fn + nb0 + c] = (__bf16)((g0v / (1.f + __expf(-g0v))) * au0[i]);
      act[r * Fn + nb1 + c] = (__bf16)((g1v / (1.f + __expf(-g1v))) * au1[i]);
    }
  }
  __syncthreads();

  // Phase 2: down GEMM over D (96 n-tiles, 12 per wave, paired), scatter-add
  for (int np = 0; np < 6; np++) {
    const int nb0 = (wave * 12 + np * 2) * 16;
    const int nb1 = nb0 + 16;
    v8f ad0 = {}, ad1 = {};
    for (int k = 0; k < Fn; k += 32) {
      v16bf a = load_A(act + k, Fn, lane);
      ad0 = wmma_bf16(a, load_B_bf16_rows(WdE + (size_t)nb0 * Fn + k, Fn, lane), ad0);
      ad1 = wmma_bf16(a, load_B_bf16_rows(WdE + (size_t)nb1 * Fn + k, Fn, lane), ad1);
    }
    const int c = lane & 15;
#pragma unroll
    for (int i = 0; i < 8; i++) {
      const int r = 8 * (lane >> 4) + i;
      atomicAdd(&Y[(size_t)sTok[r] * Dsz + nb0 + c], ad0[i] * sWt[r]);
      atomicAdd(&Y[(size_t)sTok[r] * Dsz + nb1 + c], ad1[i] * sWt[r]);
    }
  }
}

// ---------------------------------------------------------------------------
extern "C" void kernel_launch(void* const* d_in, const int* in_sizes, int n_in,
                              void* d_out, int out_size, void* d_ws, size_t ws_size,
                              hipStream_t stream) {
  const float* x   = (const float*)d_in[0];
  const float* fc  = (const float*)d_in[1];
  const float* fs  = (const float*)d_in[2];
  const float* anw = (const float*)d_in[3];
  const float* wq  = (const float*)d_in[4];
  const float* wk  = (const float*)d_in[5];
  const float* wv  = (const float*)d_in[6];
  const float* wo  = (const float*)d_in[7];
  const float* fnw = (const float*)d_in[8];
  const float* gw  = (const float*)d_in[9];
  const float* wgt = (const float*)d_in[10];
  const float* wup = (const float*)d_in[11];
  const float* wdn = (const float*)d_in[12];
  float* out = (float*)d_out;

  char* ws = (char*)d_ws;
  auto alloc = [&](size_t bytes) -> char* {
    char* p = ws; ws += (bytes + 255) & ~(size_t)255; return p;
  };
  __bf16* xn   = (__bf16*)alloc((size_t)Tn * Dsz * 2);
  __bf16* qb   = (__bf16*)alloc((size_t)Tn * HQn * HDd * 2);
  __bf16* kb   = (__bf16*)alloc((size_t)Tn * HKn * HDd * 2);
  __bf16* vb   = (__bf16*)alloc((size_t)Tn * HKn * HDd * 2);
  __bf16* ctx  = (__bf16*)alloc((size_t)Tn * HQn * HDd * 2);
  __bf16* hn   = (__bf16*)alloc((size_t)Tn * Dsz * 2);
  float*  lgts = (float*) alloc((size_t)Tn * En * 4);
  int*    tki  = (int*)   alloc((size_t)Tn * KTOP * 4);
  float*  tkw  = (float*) alloc((size_t)Tn * KTOP * 4);
  int*    cnts = (int*)   alloc(64 * 4);
  int*    tokl = (int*)   alloc((size_t)En * Tn * 4);
  float*  wtll = (float*) alloc((size_t)En * Tn * 4);
  // bf16 weight copies (MoE weights then fit in the 192MB L2)
  __bf16* wqh  = (__bf16*)alloc((size_t)HQn * HDd * Dsz * 2);
  __bf16* wkh  = (__bf16*)alloc((size_t)HKn * HDd * Dsz * 2);
  __bf16* wvh  = (__bf16*)alloc((size_t)HKn * HDd * Dsz * 2);
  __bf16* woh  = (__bf16*)alloc((size_t)Dsz * HQn * HDd * 2);
  __bf16* gwh  = (__bf16*)alloc((size_t)En * Dsz * 2);
  __bf16* wgth = (__bf16*)alloc((size_t)En * Fn * Dsz * 2);
  __bf16* wuph = (__bf16*)alloc((size_t)En * Fn * Dsz * 2);
  __bf16* wdnh = (__bf16*)alloc((size_t)En * Dsz * Fn * 2);

  // 0. weight conversion passes
  cvt_bf16_kernel<<<2048, 256, 0, stream>>>(wq, wqh, (long)HQn * HDd * Dsz);
  cvt_bf16_kernel<<<512,  256, 0, stream>>>(wk, wkh, (long)HKn * HDd * Dsz);
  cvt_bf16_kernel<<<512,  256, 0, stream>>>(wv, wvh, (long)HKn * HDd * Dsz);
  cvt_bf16_kernel<<<2048, 256, 0, stream>>>(wo, woh, (long)Dsz * HQn * HDd);
  cvt_bf16_kernel<<<96,   256, 0, stream>>>(gw, gwh, (long)En * Dsz);
  cvt_bf16_kernel<<<8192, 256, 0, stream>>>(wgt, wgth, (long)En * Fn * Dsz);
  cvt_bf16_kernel<<<8192, 256, 0, stream>>>(wup, wuph, (long)En * Fn * Dsz);
  cvt_bf16_kernel<<<8192, 256, 0, stream>>>(wdn, wdnh, (long)En * Dsz * Fn);

  // 1. attention RMSNorm
  rmsnorm_kernel<<<Tn, 256, 0, stream>>>(x, anw, xn);
  // 2. QKV projections (WMMA)
  gemm_kernel<0><<<dim3(HQn * HDd / 64, Tn / 32), 32, 0, stream>>>(xn, wqh, HQn * HDd, Dsz, qb, nullptr);
  gemm_kernel<0><<<dim3(HKn * HDd / 64, Tn / 32), 32, 0, stream>>>(xn, wkh, HKn * HDd, Dsz, kb, nullptr);
  gemm_kernel<0><<<dim3(HKn * HDd / 64, Tn / 32), 32, 0, stream>>>(xn, wvh, HKn * HDd, Dsz, vb, nullptr);
  // 3. RoPE on q and k
  {
    const int tq = Tn * HQn * 32;
    rope_kernel<<<(tq + 255) / 256, 256, 0, stream>>>(qb, HQn, fc, fs);
    const int tk = Tn * HKn * 32;
    rope_kernel<<<(tk + 255) / 256, 256, 0, stream>>>(kb, HKn, fc, fs);
  }
  // 4. causal flash attention
  attn_kernel<<<dim3(Ssz / 16, HQn, Bsz), 32, 0, stream>>>(qb, kb, vb, ctx);
  // 5. output projection + residual -> d_out holds h
  gemm_kernel<2><<<dim3(Dsz / 64, Tn / 32), 32, 0, stream>>>(ctx, woh, Dsz, HQn * HDd, out, x);
  // 6. FFN RMSNorm
  rmsnorm_kernel<<<Tn, 256, 0, stream>>>(out, fnw, hn);
  // 7. gate logits (WMMA, N=16)
  gemm16_kernel<<<dim3(En / 16, Tn / 16), 32, 0, stream>>>(hn, gwh, En, Dsz, lgts);
  // 8. top-4 routing + deterministic expert compaction
  gate_topk_kernel<<<(Tn + 255) / 256, 256, 0, stream>>>(lgts, tki, tkw);
  compact_kernel<<<1, 32, 0, stream>>>(tki, tkw, cnts, tokl, wtll);
  // 9. MoE experts (TDM token gather), accumulate into d_out (h + y)
  moe_kernel<<<dim3(Tn / 16, En), 256, 0, stream>>>(hn, wgth, wuph, wdnh, cnts, tokl, wtll, out);
}
